// TransformerDecoder_tcn_7215545058049
// MI455X (gfx1250) — compile-verified, matrix operand order checked by reading, not run on hardware
//
#include <hip/hip_runtime.h>
#include <hip/hip_bf16.h>
#include <math.h>

// ---------------------------------------------------------------------------
// GRNformer-style graph transformer for MI455X (gfx1250, wave32).
//  - Fused q/k/v/skip node GEMMs: v_wmma_f32_16x16x32_f16, B tiles staged in
//    LDS via the Tensor Data Mover (tensor_load_to_lds + s_wait_tensorcnt).
//  - Edge phase: L2-resident gathers + global f32 atomics (segment softmax).
// ---------------------------------------------------------------------------

typedef __attribute__((ext_vector_type(16))) _Float16 v16h;
typedef __attribute__((ext_vector_type(8)))  float    v8f;
typedef unsigned int v4u   __attribute__((ext_vector_type(4)));
typedef int          v8i_t __attribute__((ext_vector_type(8)));
typedef int          v4i_t __attribute__((ext_vector_type(4)));

#define NN 50000
#define NE 800000
#define NHEADS 2

struct Seg  { const float* B; const float* bias; float* C; int cols; };
struct Segs4 { Seg s[4]; int nseg; };

// ---- monotonic float<->uint key for atomicMax-based segment max ------------
__device__ __forceinline__ unsigned f32_key(float x) {
  unsigned b = __float_as_uint(x);
  return (b & 0x80000000u) ? ~b : (b | 0x80000000u);
}
__device__ __forceinline__ float key_f32(unsigned k) {
  unsigned b = (k & 0x80000000u) ? (k ^ 0x80000000u) : ~k;
  return __uint_as_float(b);
}

// ---- WMMA A fragment (ISA 7.12.2 16-bit layout), f32 global -> f16 ---------
// lane row = m0+(lane&15); halves 0-7 -> K=kb..kb+7, halves 8-15 -> kb+16..23,
// kb = k0 + (lane>=16 ? 8 : 0).
__device__ __forceinline__ v16h load_a_frag(const float* __restrict__ A, int lda,
                                            int row, int k0, int hi8) {
  const float* p = A + (size_t)row * lda + k0 + hi8;
  float t[16];
  *(float4*)&t[0]  = *(const float4*)(p + 0);
  *(float4*)&t[4]  = *(const float4*)(p + 4);
  *(float4*)&t[8]  = *(const float4*)(p + 16);
  *(float4*)&t[12] = *(const float4*)(p + 20);
  v16h f;
#pragma unroll
  for (int i = 0; i < 16; ++i) f[i] = (_Float16)t[i];
  return f;
}

// ---- WMMA B fragment from an LDS-staged K x 16 f32 tile --------------------
__device__ __forceinline__ v16h lds_b_frag(const float* ldsB, int col, int k0, int hi8) {
  int kb = k0 + hi8;
  v16h f;
#pragma unroll
  for (int i = 0; i < 8; ++i) f[i]     = (_Float16)ldsB[(kb + i) * 16 + col];
#pragma unroll
  for (int i = 0; i < 8; ++i) f[8 + i] = (_Float16)ldsB[(kb + 16 + i) * 16 + col];
  return f;
}

// ---- TDM: DMA a (tile_y x 16) f32 tile, row stride ldb, into LDS -----------
__device__ __forceinline__ void tdm_load_b_tile(unsigned lds_off, const float* gptr,
                                                int tile_y, int ldb) {
  unsigned long long ga = (unsigned long long)(uintptr_t)gptr;
  v4u g0;
  g0[0] = 1u;                                        // count=1 (valid user D#)
  g0[1] = lds_off;                                   // lds_addr (bytes)
  g0[2] = (unsigned)(ga & 0xFFFFFFFFu);              // global_addr[31:0]
  g0[3] = (unsigned)((ga >> 32) & 0x01FFFFFFu)       // global_addr[56:32]
        | (2u << 30);                                // type = 2 ("image")
  unsigned dim0 = 16u;                               // valid extent x (in-bounds tiles)
  unsigned dim1 = (unsigned)tile_y;                  // valid extent y
  unsigned long long s0 = (unsigned long long)ldb;   // row stride (elements)
  v8i_t g1;
  g1[0] = (int)(2u << 16);                           // data_size=4B, mask=0
  g1[1] = (int)((dim0 & 0xFFFFu) << 16);             // tensor_dim0[15:0]
  g1[2] = (int)(((dim0 >> 16) & 0xFFFFu) | ((dim1 & 0xFFFFu) << 16));
  g1[3] = (int)(((dim1 >> 16) & 0xFFFFu) | (16u << 16));   // tile_dim0 = 16
  g1[4] = (int)((unsigned)tile_y & 0xFFFFu);         // tile_dim1 = K, tile_dim2 = 0
  g1[5] = (int)(unsigned)(s0 & 0xFFFFFFFFu);         // dim0_stride[31:0]
  g1[6] = (int)(unsigned)((s0 >> 32) & 0xFFFFu);     // dim0_stride[47:32]
  g1[7] = 0;
  v4i_t z4 = {0, 0, 0, 0};                           // 2D tensor: groups 2/3 unused
  v8i_t z8 = {0, 0, 0, 0, 0, 0, 0, 0};
  // clang-23 / therock-10.0 form: 6 operands (extra int32x8 group, then cpol)
  __builtin_amdgcn_tensor_load_to_lds(g0, g1, z4, z4, z8, 0);
  __builtin_amdgcn_s_wait_tensorcnt(0);
}

// ---------------------------------------------------------------------------
// Fused multi-output GEMM: C_s = A @ B_s + bias_s for up to 4 segments that
// share A (M x K). One block = 8 waves = 128 rows x one 16-col tile; the
// K x 16 B tile is TDM-staged into LDS once and reused by all 8 waves.
// M % 16 == 0, every segment cols % 16 == 0, K % 32 == 0 (true for all calls).
// ---------------------------------------------------------------------------
__global__ __launch_bounds__(256) void wmma_gemm_multi(
    const float* __restrict__ A, Segs4 segs, int M, int K, int act) {
  __shared__ __align__(128) float ldsB[128 * 16];   // up to K=128

  // map blockIdx.x (global column tile) -> segment + local column offset
  int col16 = blockIdx.x * 16;
  int si = 0, base = 0;
  while (si < segs.nseg - 1 && col16 >= base + segs.s[si].cols) {
    base += segs.s[si].cols;
    ++si;
  }
  const float* Bseg  = segs.s[si].B;
  const float* bias  = segs.s[si].bias;
  float*       Cseg  = segs.s[si].C;
  const int    Nc    = segs.s[si].cols;
  const int    n0    = col16 - base;

  // stage B tile (K x 16) into LDS with the Tensor Data Mover (wave 0 only)
  if (threadIdx.x < 32) {
    unsigned lds_off = (unsigned)(uintptr_t)&ldsB[0];
    tdm_load_b_tile(lds_off, Bseg + n0, K, Nc);
  }
  __syncthreads();

  const int wave = threadIdx.x >> 5;
  const int lane = threadIdx.x & 31;
  const int hi8  = (lane >= 16) ? 8 : 0;
  const int m0   = (blockIdx.y * 8 + wave) * 16;

  if (m0 < M) {
    const int row = m0 + (lane & 15);
    const int col = lane & 15;

    v8f acc = {};
    for (int k0 = 0; k0 < K; k0 += 32) {
      v16h a = load_a_frag(A, K, row, k0, hi8);
      v16h b = lds_b_frag(ldsB, col, k0, hi8);
      acc = __builtin_amdgcn_wmma_f32_16x16x32_f16(
          false, a, false, b, (short)0, acc, false, false);
    }

    const float bb = bias ? bias[n0 + col] : 0.0f;
    const int rbase = m0 + hi8;  // C/D layout: VGPR r -> row r (+8 for hi lanes)
#pragma unroll
    for (int r = 0; r < 8; ++r) {
      float v = acc[r] + bb;
      if (act) v = fmaxf(v, 0.0f);
      Cseg[(size_t)(rbase + r) * Nc + n0 + col] = v;
    }
  }
}

// ---- edge phase ------------------------------------------------------------
__global__ void edge_alpha_kernel(const float* __restrict__ q, const float* __restrict__ kk,
                                  const float* __restrict__ ea, const float* __restrict__ we,
                                  const int* __restrict__ ei, float* __restrict__ alphaw,
                                  unsigned* __restrict__ amax, int O, int HO, float rscale) {
  int idx = blockIdx.x * blockDim.x + threadIdx.x;
  if (idx >= NE * NHEADS) return;
  int e = idx >> 1, h = idx & 1;
  int src = ei[e], dst = ei[NE + e];
  float ea0 = ea[e * NHEADS], ea1 = ea[e * NHEADS + 1];
  const float* qp = q  + (size_t)dst * HO + h * O;
  const float* kp = kk + (size_t)src * HO + h * O;
  const float* w0 = we + h * O;
  const float* w1 = we + HO + h * O;
  float s = 0.0f;
  for (int o = 0; o < O; ++o) {
    float ev = ea0 * w0[o] + ea1 * w1[o];
    s += qp[o] * (kp[o] + ev);
  }
  s *= rscale;
  alphaw[idx] = s;
  atomicMax(&amax[dst * NHEADS + h], f32_key(s));
}

__global__ void edge_exp_kernel(float* __restrict__ alphaw, const unsigned* __restrict__ amax,
                                float* __restrict__ denom, const int* __restrict__ ei) {
  int idx = blockIdx.x * blockDim.x + threadIdx.x;
  if (idx >= NE * NHEADS) return;
  int e = idx >> 1, h = idx & 1;
  int dst = ei[NE + e];
  float m  = key_f32(amax[dst * NHEADS + h]);
  float ex = __expf(alphaw[idx] - m);
  alphaw[idx] = ex;
  atomicAdd(&denom[dst * NHEADS + h], ex);
}

__global__ void edge_message_kernel(const float* __restrict__ alphaw, const float* __restrict__ denom,
                                    const float* __restrict__ vv, const float* __restrict__ ea,
                                    const float* __restrict__ we, const int* __restrict__ ei,
                                    float* __restrict__ alphan, float* __restrict__ outacc,
                                    int O, int HO) {
  int idx = blockIdx.x * blockDim.x + threadIdx.x;
  if (idx >= NE * NHEADS) return;
  int e = idx >> 1, h = idx & 1;
  int src = ei[e], dst = ei[NE + e];
  float a = alphaw[idx] / (denom[dst * NHEADS + h] + 1e-16f);
  alphan[idx] = a;
  float ea0 = ea[e * NHEADS], ea1 = ea[e * NHEADS + 1];
  const float* vp = vv + (size_t)src * HO + h * O;
  const float* w0 = we + h * O;
  const float* w1 = we + HO + h * O;
  float* op = outacc + (size_t)dst * HO + h * O;
  for (int o = 0; o < O; ++o) {
    float ev = ea0 * w0[o] + ea1 * w1[o];
    atomicAdd(&op[o], (vp[o] + ev) * a);
  }
}

__global__ void node_combine_kernel(const float* __restrict__ outacc, const float* __restrict__ skip,
                                    const float* __restrict__ wbeta, float* __restrict__ xout,
                                    int O, int HO) {
  int n = blockIdx.x * blockDim.x + threadIdx.x;
  if (n >= NN) return;
  const float* a0 = outacc + (size_t)n * HO;
  const float* a1 = a0 + O;
  const float* xr = skip + (size_t)n * O;
  float s = 0.0f;
  for (int o = 0; o < O; ++o) {
    float out_o = 0.5f * (a0[o] + a1[o]);
    float r = xr[o];
    s += out_o * wbeta[o] + r * wbeta[O + o] + (out_o - r) * wbeta[2 * O + o];
  }
  float beta = 1.0f / (1.0f + __expf(-s));
  for (int o = 0; o < O; ++o) {
    float out_o = 0.5f * (a0[o] + a1[o]);
    float r = xr[o];
    xout[(size_t)n * O + o] = beta * r + (1.0f - beta) * out_o;
  }
}

// ---- batch norm ------------------------------------------------------------
__global__ void bn_reduce_kernel(const float* __restrict__ x, float* __restrict__ sums,
                                 float* __restrict__ sqs, int C, int npb) {
  int c = threadIdx.x;
  if (c >= C) return;
  int n0 = blockIdx.x * npb;
  int n1 = n0 + npb; if (n1 > NN) n1 = NN;
  float s = 0.0f, q = 0.0f;
  for (int n = n0; n < n1; ++n) {
    float v = x[(size_t)n * C + c];
    s += v; q += v * v;
  }
  atomicAdd(&sums[c], s);
  atomicAdd(&sqs[c], q);
}

__global__ void bn_norm_kernel(const float* __restrict__ x, const float* __restrict__ sums,
                               const float* __restrict__ sqs, const float* __restrict__ gamma,
                               const float* __restrict__ beta, float* __restrict__ xn, int C) {
  int idx = blockIdx.x * blockDim.x + threadIdx.x;
  if (idx >= NN * C) return;
  int c = idx % C;
  float mean = sums[c] * (1.0f / NN);
  float var  = sqs[c] * (1.0f / NN) - mean * mean;
  float inv  = rsqrtf(var + 1e-5f);
  xn[idx] = (x[idx] - mean) * inv * gamma[c] + beta[c];
}

// ---------------------------------------------------------------------------
extern "C" void kernel_launch(void* const* d_in, const int* in_sizes, int n_in,
                              void* d_out, int out_size, void* d_ws, size_t ws_size,
                              hipStream_t stream) {
  (void)in_sizes; (void)n_in; (void)out_size; (void)ws_size;

  const float* z      = (const float*)d_in[0];
  const int*   ei     = (const int*)d_in[1];
  const float* ea_in  = (const float*)d_in[2];
  const float* fc1_w  = (const float*)d_in[33];
  const float* fc1_b  = (const float*)d_in[34];
  const float* fc2_w  = (const float*)d_in[35];
  const float* fc2_b  = (const float*)d_in[36];
  const float* bn_g   = (const float*)d_in[37];
  const float* bn_b   = (const float*)d_in[38];

  // ---- carve workspace (floats) ----
  float* base = (float*)d_ws;
  size_t off = 0;
  auto alloc = [&](size_t n) -> float* {
    float* p = base + off;
    off += (n + 255) & ~(size_t)255;
    return p;
  };
  float*    xB      = alloc((size_t)NN * 128);
  float*    xC      = alloc((size_t)NN * 128);
  float*    qb      = alloc((size_t)NN * 192);
  float*    kb      = alloc((size_t)NN * 192);
  float*    vb      = alloc((size_t)NN * 192);
  float*    skipb   = alloc((size_t)NN * 96);
  float*    outaccb = alloc((size_t)NN * 192);
  float*    alphaW  = alloc((size_t)NE * NHEADS);
  float*    alphaP0 = alloc((size_t)NE * NHEADS);
  float*    alphaP1 = alloc((size_t)NE * NHEADS);
  unsigned* amaxb   = (unsigned*)alloc((size_t)NN * NHEADS);
  float*    denomb  = alloc((size_t)NN * NHEADS);
  float*    bnsum   = alloc(256);
  float*    bnsq    = alloc(256);

  auto launch_gemm = [&](const float* A, const Segs4& sg, int total_cols,
                         int M, int K, int act) {
    dim3 grid(total_cols / 16, (M / 16 + 7) / 8);
    wmma_gemm_multi<<<grid, dim3(256), 0, stream>>>(A, sg, M, K, act);
  };

  const int IN[3] = {128, 32, 64};
  const int ON[3] = {32, 64, 96};

  const float* xin = z;
  const float* ea  = ea_in;
  float* xout = xB;
  float* alphaNext = alphaP0;

  const int ethreads = NE * NHEADS;
  const int eblocks  = (ethreads + 255) / 256;

  for (int L = 0; L < 3; ++L) {
    const float* wq    = (const float*)d_in[3 + 10 * L + 0];
    const float* bq    = (const float*)d_in[3 + 10 * L + 1];
    const float* wk    = (const float*)d_in[3 + 10 * L + 2];
    const float* bk    = (const float*)d_in[3 + 10 * L + 3];
    const float* wv    = (const float*)d_in[3 + 10 * L + 4];
    const float* bv    = (const float*)d_in[3 + 10 * L + 5];
    const float* we    = (const float*)d_in[3 + 10 * L + 6];
    const float* wskip = (const float*)d_in[3 + 10 * L + 7];
    const float* bskip = (const float*)d_in[3 + 10 * L + 8];
    const float* wbeta = (const float*)d_in[3 + 10 * L + 9];

    const int I = IN[L], O = ON[L], HO = NHEADS * O;

    // fused q/k/v/skip projection: one pass over x, WMMA + TDM-staged weights
    Segs4 sg{};
    sg.s[0] = {wq,    bq,    qb,    HO};
    sg.s[1] = {wk,    bk,    kb,    HO};
    sg.s[2] = {wv,    bv,    vb,    HO};
    sg.s[3] = {wskip, bskip, skipb, O };
    sg.nseg = 4;
    launch_gemm(xin, sg, 3 * HO + O, NN, I, 0);

    // segment accumulators
    (void)hipMemsetAsync(amaxb,   0, (size_t)NN * NHEADS * sizeof(unsigned), stream);
    (void)hipMemsetAsync(denomb,  0, (size_t)NN * NHEADS * sizeof(float),    stream);
    (void)hipMemsetAsync(outaccb, 0, (size_t)NN * HO * sizeof(float),        stream);

    const float rscale = 1.0f / sqrtf((float)O);
    edge_alpha_kernel<<<eblocks, 256, 0, stream>>>(qb, kb, ea, we, ei, alphaW, amaxb, O, HO, rscale);
    edge_exp_kernel<<<eblocks, 256, 0, stream>>>(alphaW, amaxb, denomb, ei);
    edge_message_kernel<<<eblocks, 256, 0, stream>>>(alphaW, denomb, vb, ea, we, ei,
                                                     alphaNext, outaccb, O, HO);
    node_combine_kernel<<<(NN + 255) / 256, 256, 0, stream>>>(outaccb, skipb, wbeta, xout, O, HO);

    xin = xout;
    xout = (xout == xB) ? xC : xB;
    ea = alphaNext;
    alphaNext = (alphaNext == alphaP0) ? alphaP1 : alphaP0;
  }

  // ---- batch norm + MLP head (reuse qb/kb as scratch) ----
  float* xn = qb;   // [NN, 96]
  float* h1 = kb;   // [NN, 32]
  (void)hipMemsetAsync(bnsum, 0, 96 * sizeof(float), stream);
  (void)hipMemsetAsync(bnsq,  0, 96 * sizeof(float), stream);
  bn_reduce_kernel<<<(NN + 511) / 512, 96, 0, stream>>>(xin, bnsum, bnsq, 96, 512);
  bn_norm_kernel<<<(NN * 96 + 255) / 256, 256, 0, stream>>>(xin, bnsum, bnsq, bn_g, bn_b, xn, 96);

  Segs4 g1{};
  g1.s[0] = {fc1_w, fc1_b, h1, 32};
  g1.nseg = 1;
  launch_gemm(xn, g1, 32, NN, 96, 1);   // relu

  Segs4 g2{};
  g2.s[0] = {fc2_w, fc2_b, (float*)d_out, 128};
  g2.nseg = 1;
  launch_gemm(h1, g2, 128, NN, 32, 0);  // final x -> d_out

  // tuple output: append edge_index verbatim after x [NN,128]
  (void)hipMemcpyAsync((float*)d_out + (size_t)NN * 128, d_in[1],
                       (size_t)2 * NE * sizeof(int), hipMemcpyDeviceToDevice, stream);
}
